// DilatedAttention_850403525316
// MI455X (gfx1250) — matrix operand order repaired and checked by reading, not compiled, not run on hardware
//
#include <hip/hip_runtime.h>

// ---------------------------------------------------------------------------
// MI455X (gfx1250) transformer block:
//   LN1 -> fused QKV bf16-WMMA GEMM -> dilated windowed flash-attention
//   (WMMA) -> O-proj GEMM (+resid) -> LN2 -> FFN GEMM (GELU) -> GEMM (+resid)
// All matmuls run on v_wmma_f32_16x16x32_bf16 (f32 accumulate).
// ---------------------------------------------------------------------------

typedef __attribute__((ext_vector_type(16))) __bf16 v16bf;
typedef __attribute__((ext_vector_type(8)))  __bf16 v8bf;
typedef __attribute__((ext_vector_type(8)))  float  v8f;

#define WMMA_BF16(a, b, c) \
  __builtin_amdgcn_wmma_f32_16x16x32_bf16(false, (a), false, (b), (short)0, (c), false, false)

namespace {
constexpr int Bc  = 2;
constexpr int Lc  = 4096;
constexpr int Dc  = 1024;
constexpr int Hc  = 16;
constexpr int HDc = 64;
constexpr int HIDc = 4096;
constexpr int D3  = 3 * Dc;        // fused QKV row width
constexpr int Lp  = Lc / 2;        // per-parity dilated stream length
constexpr int BL  = Bc * Lc;       // 8192 rows
}

// Load a 16-element bf16 fragment from two contiguous 8-element (16B) chunks.
__device__ __forceinline__ v16bf ld_frag16(const __bf16* p0, const __bf16* p1) {
  v8bf a = *(const v8bf*)p0;
  v8bf b = *(const v8bf*)p1;
  v16bf r;
#pragma unroll
  for (int i = 0; i < 8; ++i) { r[i] = a[i]; r[8 + i] = b[i]; }
  return r;
}

// ---------------------------------------------------------------------------
// LayerNorm (f32 in) -> bf16 out. One 256-thread block per row (D=1024).
// ---------------------------------------------------------------------------
__global__ __launch_bounds__(256) void layernorm_to_bf16(
    const float* __restrict__ x, const float* __restrict__ g,
    const float* __restrict__ be, __bf16* __restrict__ out) {
  const int t = threadIdx.x;
  const size_t row = blockIdx.x;
  const float* xr = x + row * Dc;
  float vals[4];
  float s = 0.f, s2 = 0.f;
#pragma unroll
  for (int i = 0; i < 4; ++i) {
    float v = xr[t + 256 * i];
    vals[i] = v; s += v; s2 += v * v;
  }
#pragma unroll
  for (int d = 16; d >= 1; d >>= 1) {
    s  += __shfl_xor(s,  d, 32);
    s2 += __shfl_xor(s2, d, 32);
  }
  __shared__ float as_[8], as2_[8];
  if ((t & 31) == 0) { as_[t >> 5] = s; as2_[t >> 5] = s2; }
  __syncthreads();
  float ts = 0.f, ts2 = 0.f;
#pragma unroll
  for (int w = 0; w < 8; ++w) { ts += as_[w]; ts2 += as2_[w]; }
  const float mu   = ts * (1.f / Dc);
  const float var  = ts2 * (1.f / Dc) - mu * mu;
  const float rstd = rsqrtf(var + 1e-5f);
  __bf16* orow = out + row * Dc;
#pragma unroll
  for (int i = 0; i < 4; ++i) {
    int c = t + 256 * i;
    orow[c] = (__bf16)((vals[i] - mu) * rstd * g[c] + be[c]);
  }
}

// ---------------------------------------------------------------------------
// Weight / bias packing into bf16 / f32 workspace (done per-launch, no state).
// ---------------------------------------------------------------------------
__global__ void pack_weight_bf16(const float* __restrict__ src, __bf16* __restrict__ dst,
                                 int K, int N, int dstld, int col0) {
  size_t i = (size_t)blockIdx.x * 256 + threadIdx.x;
  if (i >= (size_t)K * N) return;
  int k = (int)(i / N), n = (int)(i % N);
  dst[(size_t)k * dstld + col0 + n] = (__bf16)src[i];
}

__global__ void pack_f32(const float* __restrict__ src, float* __restrict__ dst,
                         int n, int off) {
  int i = blockIdx.x * 256 + threadIdx.x;
  if (i < n) dst[off + i] = src[i];
}

// ---------------------------------------------------------------------------
// bf16 WMMA GEMM: C[M,N] = epilogue(A[M,K] @ W[K,N] + bias (+resid))
// Workgroup tile 128x64, 8 waves (4x2), each wave 32x32 via 2x2 WMMA frags.
// EPI: 0 = bf16 out, 1 = f32 out + residual, 2 = GELU -> bf16 out.
// ---------------------------------------------------------------------------
template <int EPI>
__global__ __launch_bounds__(256, 2) void gemm_bf16_wmma(
    const __bf16* __restrict__ A, const __bf16* __restrict__ W,
    const float* __restrict__ bias, const float* __restrict__ resid,
    void* __restrict__ outp, int M, int N, int K, int lda, int ldw) {
  __shared__ __align__(16) __bf16 As[128][40];  // +8 pad vs LDS banks
  __shared__ __align__(16) __bf16 Ws[32][72];

  const int m0   = blockIdx.y * 128;
  const int n0   = blockIdx.x * 64;
  const int t    = threadIdx.x;
  const int lane = t & 31;
  const int wave = t >> 5;
  const int wm   = wave >> 1;            // 0..3 -> M sub-tile
  const int wn   = wave & 1;             // 0..1 -> N sub-tile
  const int hn   = lane & 15;
  const int kb8  = (lane >> 4) * 8;      // A-frag K base (pairs)
  const int kb16 = (lane >> 4) * 16;     // unused; B-frag K base is kb8 too
  const int mb   = (lane >> 4) * 8;      // C-layout M base

  v8f acc[2][2] = {};

  for (int k0 = 0; k0 < K; k0 += 32) {
    __syncthreads();
    // Stage A tile (128x32): 2 x 16B chunks per thread.
#pragma unroll
    for (int c = 0; c < 2; ++c) {
      int chunk = t + 256 * c;
      int r = chunk >> 2;
      int col = (chunk & 3) * 8;
      const __bf16* gp = A + (size_t)(m0 + r) * lda + k0 + col;
      *(float4*)(&As[r][col]) = *(const float4*)gp;
      if (k0 + 32 < K) __builtin_prefetch(gp + 32, 0, 0);  // global_prefetch_b8
    }
    // Stage W tile (32x64): 1 x 16B chunk per thread.
    {
      int r = t >> 3;
      int col = (t & 7) * 8;
      const __bf16* gp = W + (size_t)(k0 + r) * ldw + n0 + col;
      *(float4*)(&Ws[r][col]) = *(const float4*)gp;
      if (k0 + 32 < K) __builtin_prefetch(gp + (size_t)32 * ldw, 0, 0);
    }
    __syncthreads();

    const int mr0 = wm * 32 + hn;
    const int mr1 = mr0 + 16;
    v16bf a0 = ld_frag16(&As[mr0][kb8], &As[mr0][16 + kb8]);
    v16bf a1 = ld_frag16(&As[mr1][kb8], &As[mr1][16 + kb8]);
    const int nc0 = wn * 32 + hn;
    const int nc1 = nc0 + 16;
    v16bf b0, b1;
#pragma unroll
    for (int e = 0; e < 16; ++e) {
      b0[e] = Ws[kb8 + (e < 8 ? e : 8 + e)][nc0];
      b1[e] = Ws[kb8 + (e < 8 ? e : 8 + e)][nc1];
    }
    acc[0][0] = WMMA_BF16(a0, b0, acc[0][0]);
    acc[0][1] = WMMA_BF16(a0, b1, acc[0][1]);
    acc[1][0] = WMMA_BF16(a1, b0, acc[1][0]);
    acc[1][1] = WMMA_BF16(a1, b1, acc[1][1]);
  }

#pragma unroll
  for (int i = 0; i < 2; ++i)
#pragma unroll
    for (int j = 0; j < 2; ++j) {
      const int gm_base = m0 + wm * 32 + 16 * i + mb;
      const int gn = n0 + wn * 32 + 16 * j + hn;
      const float bval = bias ? bias[gn] : 0.f;
#pragma unroll
      for (int r = 0; r < 8; ++r) {
        size_t idx = (size_t)(gm_base + r) * N + gn;
        float c = acc[i][j][r] + bval;
        if constexpr (EPI == 1) {
          ((float*)outp)[idx] = c + resid[idx];
        } else if constexpr (EPI == 2) {
          float gl = 0.5f * c * (1.f + erff(c * 0.70710678118f));  // exact GELU
          ((__bf16*)outp)[idx] = (__bf16)gl;
        } else {
          ((__bf16*)outp)[idx] = (__bf16)c;
        }
      }
    }
}

// ---------------------------------------------------------------------------
// Dilated windowed attention. Parity-p positions form a length-2048 stream
// where the mask is a causal sliding window of 129 steps. One wave handles
// one (batch, head, parity, 16-query tile); 5 fixed 32-key steps with online
// softmax. QK^T and P@V both via v_wmma_f32_16x16x32_bf16.
// qkv layout: [B, L, {Q|K|V} x 1024] row stride 3072; ctx: [B, L, 1024] bf16.
// ---------------------------------------------------------------------------
__global__ __launch_bounds__(128, 4) void dilated_attn_wmma(
    const __bf16* __restrict__ qkv, __bf16* __restrict__ ctx) {
  __shared__ __align__(16) __bf16 Pst[4][16][32];  // per-wave P staging

  const int lane = threadIdx.x & 31;
  const int wave = threadIdx.x >> 5;
  const int task = blockIdx.x * 4 + wave;           // 8192 tasks
  const int ntile = task & 127;
  const int par   = (task >> 7) & 1;
  const int head  = (task >> 8) & 15;
  const int batch = task >> 12;
  const int t0 = ntile * 16;

  const int hn   = lane & 15;
  const int mb   = (lane >> 4) * 8;
  const int kb8  = (lane >> 4) * 8;
  const int kb16 = (lane >> 4) * 16;

  const __bf16* base = qkv + (size_t)batch * Lc * D3 + head * HDc;
  const __bf16* Kb = base + Dc;
  const __bf16* Vb = base + 2 * Dc;

  // Q A-fragments (row m = lane&15 -> query t0+m), HD=64 -> two K-chunks.
  v16bf qf0, qf1;
  {
    const int rawq = 2 * (t0 + hn) + par;
    const __bf16* qrow = base + (size_t)rawq * D3;
    qf0 = ld_frag16(qrow + kb8,      qrow + 16 + kb8);
    qf1 = ld_frag16(qrow + 32 + kb8, qrow + 48 + kb8);
  }

  v8f o[4] = {};
  float mrow[8], lrow[8];
#pragma unroll
  for (int r = 0; r < 8; ++r) { mrow[r] = -1e30f; lrow[r] = 0.f; }

  for (int s = 0; s < 5; ++s) {
    const int kstart = t0 - 128 + 32 * s;

    // ---- scores: S = Q @ K^T (16 queries x 32 keys) ----
    v8f sc[2];
#pragma unroll
    for (int jt = 0; jt < 2; ++jt) {
      int tk = kstart + jt * 16 + hn;                 // key column for this lane
      int tkc = tk < 0 ? 0 : (tk > Lp - 1 ? Lp - 1 : tk);
      const __bf16* krow = Kb + (size_t)(2 * tkc + par) * D3;
      v16bf k0f = ld_frag16(krow + kb8,      krow + 16 + kb8);
      v16bf k1f = ld_frag16(krow + 32 + kb8, krow + 48 + kb8);
      v8f z = {};
      z = WMMA_BF16(qf0, k0f, z);
      z = WMMA_BF16(qf1, k1f, z);
      sc[jt] = z;
    }

    // ---- mask + online softmax (rows live in 16-lane halves) ----
    float pv0[8], pv1[8];
#pragma unroll
    for (int r = 0; r < 8; ++r) {
      const int tq  = t0 + mb + r;
      const int tk0 = kstart + hn;
      const int tk1 = tk0 + 16;
      const bool v0ok = (tk0 >= 0) & (tk0 <= tq) & (tq - tk0 <= 128);
      const bool v1ok = (tk1 >= 0) & (tk1 <= tq) & (tq - tk1 <= 128);
      float s0 = v0ok ? sc[0][r] * 0.125f : -1e30f;   // 1/sqrt(64)
      float s1 = v1ok ? sc[1][r] * 0.125f : -1e30f;
      float mx = fmaxf(s0, s1);
#pragma unroll
      for (int d = 8; d >= 1; d >>= 1) mx = fmaxf(mx, __shfl_xor(mx, d, 32));
      const float mnew = fmaxf(mrow[r], mx);
      const float fac = __expf(mrow[r] - mnew);
      const float p0 = v0ok ? __expf(s0 - mnew) : 0.f;
      const float p1 = v1ok ? __expf(s1 - mnew) : 0.f;
      float sum = p0 + p1;
#pragma unroll
      for (int d = 8; d >= 1; d >>= 1) sum += __shfl_xor(sum, d, 32);
      lrow[r] = lrow[r] * fac + sum;
      mrow[r] = mnew;
#pragma unroll
      for (int n2 = 0; n2 < 4; ++n2) o[n2][r] *= fac;
      pv0[r] = p0; pv1[r] = p1;
    }

    // ---- relayout P (C-layout -> A-layout) through per-wave LDS ----
    __syncthreads();
#pragma unroll
    for (int r = 0; r < 8; ++r) {
      Pst[wave][mb + r][hn]      = (__bf16)pv0[r];
      Pst[wave][mb + r][16 + hn] = (__bf16)pv1[r];
    }
    __syncthreads();
    const __bf16* prow = &Pst[wave][hn][0];
    v16bf pf = ld_frag16(prow + kb8, prow + 16 + kb8);

    // ---- O += P @ V (32 keys x 64 dims -> 4 WMMA) ----
#pragma unroll
    for (int n2 = 0; n2 < 4; ++n2) {
      v16bf vf;
#pragma unroll
      for (int e = 0; e < 16; ++e) {
        int tk = kstart + kb8 + (e < 8 ? e : 8 + e);
        int tkc = tk < 0 ? 0 : (tk > Lp - 1 ? Lp - 1 : tk);
        vf[e] = Vb[(size_t)(2 * tkc + par) * D3 + n2 * 16 + hn];
      }
      o[n2] = WMMA_BF16(pf, vf, o[n2]);
    }
  }

  // ---- normalize + store ctx (bf16) ----
#pragma unroll
  for (int r = 0; r < 8; ++r) {
    const float inv = 1.f / lrow[r];
    const int rawi = 2 * (t0 + mb + r) + par;
    __bf16* crow = ctx + ((size_t)batch * Lc + rawi) * Dc + head * HDc;
#pragma unroll
    for (int n2 = 0; n2 < 4; ++n2)
      crow[n2 * 16 + hn] = (__bf16)(o[n2][r] * inv);
  }
}

// ---------------------------------------------------------------------------
// Host-side orchestration. Workspace need ~217 MiB.
// ---------------------------------------------------------------------------
extern "C" void kernel_launch(void* const* d_in, const int* in_sizes, int n_in,
                              void* d_out, int out_size, void* d_ws, size_t ws_size,
                              hipStream_t stream) {
  (void)in_sizes; (void)n_in; (void)out_size; (void)ws_size;

  const float* x     = (const float*)d_in[0];
  const float* ln1_g = (const float*)d_in[1];
  const float* ln1_b = (const float*)d_in[2];
  const float* wq    = (const float*)d_in[3];
  const float* wk    = (const float*)d_in[4];
  const float* wv    = (const float*)d_in[5];
  const float* bq    = (const float*)d_in[6];
  const float* bk    = (const float*)d_in[7];
  const float* bv    = (const float*)d_in[8];
  const float* wo    = (const float*)d_in[9];
  const float* bo    = (const float*)d_in[10];
  const float* ln2_g = (const float*)d_in[11];
  const float* ln2_b = (const float*)d_in[12];
  const float* w1    = (const float*)d_in[13];
  const float* b1    = (const float*)d_in[14];
  const float* w2    = (const float*)d_in[15];
  const float* b2    = (const float*)d_in[16];

  char* wsb = (char*)d_ws;
  size_t off = 0;
  auto take = [&](size_t bytes) -> void* {
    void* p = wsb + off;
    off += (bytes + 255) & ~(size_t)255;
    return p;
  };
  const size_t BLD = (size_t)BL * Dc;
  __bf16* wqkv_b = (__bf16*)take((size_t)3 * Dc * Dc * 2);   // [1024][3072]
  __bf16* wo_b   = (__bf16*)take((size_t)Dc * Dc * 2);       // [1024][1024]
  __bf16* w1_b   = (__bf16*)take((size_t)Dc * HIDc * 2);     // [1024][4096]
  __bf16* w2_b   = (__bf16*)take((size_t)HIDc * Dc * 2);     // [4096][1024]
  float*  bqkv   = (float*) take((size_t)D3 * 4);            // packed q|k|v bias
  __bf16* hbuf   = (__bf16*)take(BLD * 2);                   // LN1 out
  __bf16* qkvbuf = (__bf16*)take(BLD * 3 * 2);               // fused QKV
  __bf16* ctxbuf = (__bf16*)take(BLD * 2);                   // attention out
  float*  x1     = (float*) take(BLD * 4);                   // x + attn
  __bf16* h2buf  = (__bf16*)take(BLD * 2);                   // LN2 out
  __bf16* ffn1   = (__bf16*)take((size_t)BL * HIDc * 2);     // gelu(h@w1+b1)

  const dim3 blk(256);
  // Pack weights/biases to bf16/f32 workspace.
  pack_weight_bf16<<<(Dc * Dc + 255) / 256, blk, 0, stream>>>(wq, wqkv_b, Dc, Dc, D3, 0);
  pack_weight_bf16<<<(Dc * Dc + 255) / 256, blk, 0, stream>>>(wk, wqkv_b, Dc, Dc, D3, Dc);
  pack_weight_bf16<<<(Dc * Dc + 255) / 256, blk, 0, stream>>>(wv, wqkv_b, Dc, Dc, D3, 2 * Dc);
  pack_weight_bf16<<<(Dc * Dc + 255) / 256, blk, 0, stream>>>(wo, wo_b, Dc, Dc, Dc, 0);
  pack_weight_bf16<<<(Dc * HIDc + 255) / 256, blk, 0, stream>>>(w1, w1_b, Dc, HIDc, HIDc, 0);
  pack_weight_bf16<<<(HIDc * Dc + 255) / 256, blk, 0, stream>>>(w2, w2_b, HIDc, Dc, Dc, 0);
  pack_f32<<<(Dc + 255) / 256, blk, 0, stream>>>(bq, bqkv, Dc, 0);
  pack_f32<<<(Dc + 255) / 256, blk, 0, stream>>>(bk, bqkv, Dc, Dc);
  pack_f32<<<(Dc + 255) / 256, blk, 0, stream>>>(bv, bqkv, Dc, 2 * Dc);

  // LN1
  layernorm_to_bf16<<<BL, blk, 0, stream>>>(x, ln1_g, ln1_b, hbuf);
  // Fused QKV GEMM: [8192,1024] @ [1024,3072] -> bf16
  gemm_bf16_wmma<0><<<dim3(D3 / 64, BL / 128), blk, 0, stream>>>(
      hbuf, wqkv_b, bqkv, nullptr, qkvbuf, BL, D3, Dc, Dc, D3);
  // Dilated windowed attention (WMMA flash): 8192 wave-tasks, 4 waves/block
  dilated_attn_wmma<<<(Bc * Hc * 2 * (Lp / 16)) / 4, dim3(128), 0, stream>>>(qkvbuf, ctxbuf);
  // O projection + residual: x1 = x + ctx @ wo + bo (f32)
  gemm_bf16_wmma<1><<<dim3(Dc / 64, BL / 128), blk, 0, stream>>>(
      ctxbuf, wo_b, bo, x, x1, BL, Dc, Dc, Dc, Dc);
  // LN2
  layernorm_to_bf16<<<BL, blk, 0, stream>>>(x1, ln2_g, ln2_b, h2buf);
  // FFN1 with exact GELU -> bf16
  gemm_bf16_wmma<2><<<dim3(HIDc / 64, BL / 128), blk, 0, stream>>>(
      h2buf, w1_b, b1, nullptr, ffn1, BL, HIDc, Dc, Dc, HIDc);
  // FFN2 + residual -> final f32 output
  gemm_bf16_wmma<1><<<dim3(Dc / 64, BL / 128), blk, 0, stream>>>(
      ffn1, w2_b, b2, x1, (float*)d_out, BL, Dc, HIDc, HIDc, Dc);
}
